// SQConv1d_85684597555511
// MI455X (gfx1250) — 8-bit matrix operand order repaired and checked by reading, compiled, not run on hardware
//
#include <hip/hip_runtime.h>
#include <hip/hip_bf16.h>

typedef __attribute__((ext_vector_type(8))) int v8i;

#define BS   8
#define IC   64
#define LEN  32768
#define OC   64
#define KW   3
#define NCOL (IC * KW)
#define QMAXF 127.0f
#define MAGIC 12582912.0f


__device__ __forceinline__ unsigned pack4_lowbytes(unsigned q0, unsigned q1,
                                                   unsigned q2, unsigned q3) {
#if __has_builtin(__builtin_amdgcn_perm)
    unsigned lo = __builtin_amdgcn_perm(q1, q0, 0x06050400u);
    unsigned hi = __builtin_amdgcn_perm(q3, q2, 0x06050400u);
    return __builtin_amdgcn_perm(hi, lo, 0x05040100u);
#else
    return (q0 & 0xFFu) | ((q1 & 0xFFu) << 8) | ((q2 & 0xFFu) << 16) | (q3 << 24);
#endif
}

__global__ void sq_init_kernel(unsigned* act_bits) {
    int t = blockIdx.x * blockDim.x + threadIdx.x;
    if (t < NCOL) act_bits[t] = 0u;
}

__global__ __launch_bounds__(256) void sq_act_scale_kernel(const float* __restrict__ x,
                                                           unsigned* __restrict__ act_bits) {
    int b  = blockIdx.x / IC;
    int ic = blockIdx.x % IC;
    const float4* xp4 = (const float4*)(x + ((size_t)b * IC + ic) * LEN);
    float m_all = 0.f, m_nf = 0.f, m_nl = 0.f;
    for (int i4 = threadIdx.x; i4 < LEN / 4; i4 += 256) {
        float4 v = xp4[i4];
        float a0 = fabsf(v.x), a1 = fabsf(v.y), a2 = fabsf(v.z), a3 = fabsf(v.w);
        float mi = fmaxf(fmaxf(a0, a1), fmaxf(a2, a3));
        m_all = fmaxf(m_all, mi);
        float nf = (i4 == 0) ? fmaxf(fmaxf(a1, a2), a3) : mi;
        float nl = (i4 == LEN / 4 - 1) ? fmaxf(fmaxf(a0, a1), a2) : mi;
        m_nf = fmaxf(m_nf, nf);
        m_nl = fmaxf(m_nl, nl);
    }
    __shared__ float sh[3][256];
    sh[0][threadIdx.x] = m_nl; sh[1][threadIdx.x] = m_all; sh[2][threadIdx.x] = m_nf;
    __syncthreads();
    for (int s = 128; s > 0; s >>= 1) {
        if (threadIdx.x < (unsigned)s)
            for (int v = 0; v < 3; ++v)
                sh[v][threadIdx.x] = fmaxf(sh[v][threadIdx.x], sh[v][threadIdx.x + s]);
        __syncthreads();
    }
    if (threadIdx.x < 3)
        atomicMax(&act_bits[ic * 3 + threadIdx.x], __float_as_uint(sh[threadIdx.x][0]));
}

__global__ __launch_bounds__(256) void sq_prep_kernel(const float* __restrict__ w,
                                                      const unsigned* __restrict__ act_bits,
                                                      float* __restrict__ inv_scale_sx,
                                                      float* __restrict__ sxsw_out,
                                                      unsigned* __restrict__ pbB) {
    __shared__ float sh_scale[NCOL];
    __shared__ float sh_ax[256], sh_wx[256];
    __shared__ float sh_sx, sh_sw;
    int t = threadIdx.x;
    float ax = 0.f, wx = 0.f;
    if (t < NCOL) {
        float wmax = 0.f;
        for (int oc = 0; oc < OC; ++oc) wmax = fmaxf(wmax, fabsf(w[oc * NCOL + t]));
        float act   = __uint_as_float(act_bits[t]);
        float scale = sqrtf(act) / sqrtf(wmax);
        if (scale == 0.f) scale = 1.f;
        sh_scale[t] = scale;
        ax = act / scale;
        wx = wmax * scale;
    }
    sh_ax[t] = ax; sh_wx[t] = wx;
    __syncthreads();
    for (int s = 128; s > 0; s >>= 1) {
        if (t < s) { sh_ax[t] = fmaxf(sh_ax[t], sh_ax[t + s]);
                     sh_wx[t] = fmaxf(sh_wx[t], sh_wx[t + s]); }
        __syncthreads();
    }
    if (t == 0) {
        float amax_x = sh_ax[0], amax_w = sh_wx[0];
        sh_sx = (amax_x == 0.f) ? 1.f : amax_x / QMAXF;
        sh_sw = (amax_w == 0.f) ? 1.f : amax_w / QMAXF;
        *sxsw_out = sh_sx * sh_sw;
    }
    __syncthreads();
    if (t < NCOL) inv_scale_sx[t] = 1.f / (sh_scale[t] * sh_sx);
    float inv_sw = 1.f / sh_sw;

    for (int idx = t; idx < 3072; idx += blockDim.x) {
        int v    = idx & 7;
        int lane = (idx >> 3) & 31;
        int nt   = (idx >> 8) & 3;
        int kc   = idx >> 10;
        int n    = lane & 15, half = lane >> 4;
        int oc   = nt * 16 + n;
        unsigned d = 0;
        #pragma unroll
        for (int b2 = 0; b2 < 4; ++b2) {
            int K   = (v < 4) ? (half * 16 + v * 4 + b2)
                              : (32 + half * 16 + (v - 4) * 4 + b2);
            int col = kc * 64 + K;
            float q = rintf(w[oc * NCOL + col] * sh_scale[col] * inv_sw);
            q = fminf(fmaxf(q, -QMAXF), QMAXF);
            d |= ((unsigned)(((int)q) & 0xFF)) << (8 * b2);
        }
        pbB[idx] = d;
    }
}

__global__ __launch_bounds__(256) void sq_gemm_kernel(const float* __restrict__ x,
                                                      const float* __restrict__ bias,
                                                      const float* __restrict__ inv_scale_sx,
                                                      const float* __restrict__ sxsw_p,
                                                      const unsigned* __restrict__ pbB,
                                                      float* __restrict__ out) {
    constexpr int MBLK     = 128;
    constexpr int XSTRIDE  = 136;
    constexpr int AQSTRIDE = 200;

    __shared__ __align__(16) unsigned char pool[IC * XSTRIDE * 4];
    __shared__ unsigned bsh[3072];
    __shared__ float    sc[NCOL];
    __shared__ float    bvals[OC];

    float*         xs  = (float*)pool;
    unsigned char* aqb = (unsigned char*)pool;

    int b    = blockIdx.x >> 8;
    int mblk = blockIdx.x & 255;
    int pos0 = mblk * MBLK;
    int tid  = threadIdx.x;

    __builtin_prefetch(pbB, 0, 0);

    const float* xb = x + (size_t)b * IC * LEN;
    for (int idx = tid; idx < IC * (XSTRIDE / 4); idx += 256) {
        int ic = idx / (XSTRIDE / 4);
        int i4 = idx - ic * (XSTRIDE / 4);
        int p  = pos0 - 4 + i4 * 4;
        float4 v = make_float4(0.f, 0.f, 0.f, 0.f);
        if (p >= 0 && p <= LEN - 4)
            v = *(const float4*)(xb + (size_t)ic * LEN + p);
        *(float4*)(xs + ic * XSTRIDE + i4 * 4) = v;
    }
    for (int idx = tid; idx < 3072; idx += 256) bsh[idx] = pbB[idx];
    if (tid < NCOL) sc[tid] = inv_scale_sx[tid];
    if (tid < OC)   bvals[tid] = bias[tid];
    float sxsw = *sxsw_p;
    __syncthreads();

    int m     = tid & 127;
    int cpart = tid >> 7;
    const float* xbase  = xs + (cpart * 32) * XSTRIDE + m + 3;
    const float* scbase = sc + cpart * 96;
    unsigned packed[24];
    #pragma unroll
    for (int g = 0; g < 8; ++g) {
        #pragma unroll
        for (int d = 0; d < 3; ++d) {
            unsigned qb[4];
            #pragma unroll
            for (int j = 0; j < 4; ++j) {
                const int c  = g * 12 + d * 4 + j;
                const int io = c / 3;
                const int kk = c % 3;
                float t = fmaf(xbase[io * XSTRIDE + kk], scbase[c], MAGIC);
                t = fminf(fmaxf(t, MAGIC - QMAXF), MAGIC + QMAXF);
                qb[j] = __float_as_uint(t);
            }
            packed[g * 3 + d] = pack4_lowbytes(qb[0], qb[1], qb[2], qb[3]);
        }
    }
    __syncthreads();

    {
        unsigned* aqrow = (unsigned*)(aqb + m * AQSTRIDE + cpart * 96);
        #pragma unroll
        for (int i = 0; i < 24; ++i) aqrow[i] = packed[i];
    }
    __syncthreads();

    int wave = tid >> 5;
    int lane = tid & 31;
    int row  = lane & 15;
    int half = lane >> 4;
    int mrow = wave * 16 + row;

    v8i acc[4];
    #pragma unroll
    for (int nt = 0; nt < 4; ++nt)
        #pragma unroll
        for (int r = 0; r < 8; ++r) acc[nt][r] = 0;

    const unsigned char* ab = aqb + (size_t)mrow * AQSTRIDE + (half << 4);
    #pragma unroll
    for (int kc = 0; kc < 3; ++kc) {
        const int col0 = kc * 64;
        v8i a;
        {
            uint2 p01 = *(const uint2*)(ab + col0 + 0);
            uint2 p23 = *(const uint2*)(ab + col0 + 8);
            uint2 p45 = *(const uint2*)(ab + col0 + 32);
            uint2 p67 = *(const uint2*)(ab + col0 + 40);
            a[0] = (int)p01.x; a[1] = (int)p01.y;
            a[2] = (int)p23.x; a[3] = (int)p23.y;
            a[4] = (int)p45.x; a[5] = (int)p45.y;
            a[6] = (int)p67.x; a[7] = (int)p67.y;
        }
        #pragma unroll
        for (int nt = 0; nt < 4; ++nt) {
            const uint4* bp = (const uint4*)&bsh[(((kc * 4 + nt) * 32) + lane) * 8];
            uint4 q0 = bp[0], q1 = bp[1];
            v8i bf;
            bf[0] = (int)q0.x; bf[1] = (int)q0.y; bf[2] = (int)q0.z; bf[3] = (int)q0.w;
            bf[4] = (int)q1.x; bf[5] = (int)q1.y; bf[6] = (int)q1.z; bf[7] = (int)q1.w;
            acc[nt] = __builtin_amdgcn_wmma_i32_16x16x64_iu8(
                true, a, true, bf, acc[nt], false, false);
        }
    }

    #pragma unroll
    for (int nt = 0; nt < 4; ++nt) {
        int oc   = nt * 16 + (lane & 15);
        int posb = pos0 + wave * 16 + half * 8;
        float bb = bvals[oc];
        float vals[8];
        #pragma unroll
        for (int r = 0; r < 8; ++r) vals[r] = (float)acc[nt][r] * sxsw + bb;
        float* dst = out + ((size_t)b * OC + oc) * LEN + posb;
        *(float4*)(dst)     = make_float4(vals[0], vals[1], vals[2], vals[3]);
        *(float4*)(dst + 4) = make_float4(vals[4], vals[5], vals[6], vals[7]);
    }
}

extern "C" void kernel_launch(void* const* d_in, const int* in_sizes, int n_in,
                              void* d_out, int out_size, void* d_ws, size_t ws_size,
                              hipStream_t stream) {
    const float* x      = (const float*)d_in[0];
    const float* weight = (const float*)d_in[1];
    const float* bias   = (const float*)d_in[2];
    float* out = (float*)d_out;

    unsigned* act_bits     = (unsigned*)d_ws;
    float*    inv_scale_sx = (float*)((char*)d_ws + 1024);
    float*    sxsw         = (float*)((char*)d_ws + 2048);
    unsigned* pbB          = (unsigned*)((char*)d_ws + 4096);

    sq_init_kernel<<<1, 256, 0, stream>>>(act_bits);
    sq_act_scale_kernel<<<BS * IC, 256, 0, stream>>>(x, act_bits);
    sq_prep_kernel<<<1, 256, 0, stream>>>(weight, act_bits, inv_scale_sx, sxsw, pbB);
    sq_gemm_kernel<<<BS * (LEN / 128), 256, 0, stream>>>(x, bias, inv_scale_sx, sxsw, pbB, out);
}
